// Informer_39625368273642
// MI455X (gfx1250) — matrix operand order repaired and checked by reading, not compiled, not run on hardware
//
#include <hip/hip_runtime.h>
#include <math.h>

#define B_      8
#define S_      1024
#define ENCIN_  3
#define D_      512
#define H_      8
#define DH_     64
#define DFF_    2048
#define NL_     3
#define NTOP_   35
#define NCLS_   5
#define EPS_    1e-5f

typedef __attribute__((ext_vector_type(16))) _Float16 v16h;
typedef __attribute__((ext_vector_type(8)))  float    v8f;

// ---- Tensor Data Mover availability (this toolchain: 6-arg builtin) --------
#if defined(__gfx1250__) && defined(__has_builtin)
#if __has_builtin(__builtin_amdgcn_tensor_load_to_lds) && \
    __has_builtin(__builtin_amdgcn_s_wait_tensorcnt)
#define USE_TDM 1
#endif
#endif
#ifndef USE_TDM
#define USE_TDM 0
#endif

typedef unsigned int u32;
typedef u32 u32x4 __attribute__((ext_vector_type(4)));
typedef int i32x4 __attribute__((ext_vector_type(4)));
typedef int i32x8 __attribute__((ext_vector_type(8)));

#if USE_TDM
// 2D f16 tile load, global -> LDS, via TDM descriptor (ISA ch.8 D# layout).
// tileX elements contiguous (K), tileY rows, row stride = strideElems.
__device__ __forceinline__
void tdm_load_tile(const _Float16* gsrc, u32 ldsOff,
                   u32 tileX, u32 tileY, u32 strideElems)
{
    unsigned long long ga = (unsigned long long)(uintptr_t)gsrc;
    u32x4 g0;
    g0[0] = 1u;                                            // count=1 (valid)
    g0[1] = ldsOff;                                        // lds_addr (bytes)
    g0[2] = (u32)ga;                                       // global_addr[31:0]
    g0[3] = (u32)((ga >> 32) & 0x01FFFFFFu) | (2u << 30);  // [56:32] + type=2
    i32x8 g1;
    g1[0] = (int)(1u << 16);                 // workgroup_mask=0, data_size=1 (2B)
    g1[1] = (int)((tileX & 0xFFFFu) << 16);  // tensor_dim0[15:0] (==tile: no clip)
    g1[2] = (int)((tileY & 0xFFFFu) << 16);  // tensor_dim0[31:16]=0 | tensor_dim1[15:0]
    g1[3] = (int)((tileX & 0xFFFFu) << 16);  // tensor_dim1[31:16]=0 | tile_dim0
    g1[4] = (int)(tileY & 0xFFFFu);          // tile_dim1 | tile_dim2=0
    g1[5] = (int)strideElems;                // tensor_dim0_stride[31:0]
    g1[6] = 0;                               // stride[47:32]=0 | dim1_stride[15:0]=0
    g1[7] = 0;
    i32x4 z4 = {0, 0, 0, 0};
    i32x8 z8 = {0, 0, 0, 0, 0, 0, 0, 0};
    // 6-arg form (clang-23 / therock-10.0 headers): groups 2/3 disabled (2D).
    __builtin_amdgcn_tensor_load_to_lds(g0, g1, z4, z4, z8, 0);
}
#endif

// ---------------------------------------------------------------------------
// fp32 -> f16 conversion (weights once per layer, activations once per GEMM)
// ---------------------------------------------------------------------------
__global__ void f32_to_f16_kernel(const float* __restrict__ in,
                                  _Float16* __restrict__ out, int n)
{
    int i = blockIdx.x * blockDim.x + threadIdx.x;
    if (i < n) out[i] = (_Float16)in[i];
}

// ---------------------------------------------------------------------------
// Token embedding: circular conv1d (W=3, Cin=3, Cout=512) + positional emb
// ---------------------------------------------------------------------------
__global__ void token_embed_kernel(const float* __restrict__ x_enc,
                                   const float* __restrict__ tk,
                                   float* __restrict__ X)
{
    int idx = blockIdx.x * blockDim.x + threadIdx.x;
    if (idx >= B_ * S_ * D_) return;
    int dm = idx % D_;
    int s  = (idx / D_) % S_;
    int b  = idx / (D_ * S_);

    float acc = 0.f;
    #pragma unroll
    for (int w = 0; w < 3; ++w) {
        int j = s - 1 + w;
        j = (j < 0) ? (S_ - 1) : ((j >= S_) ? 0 : j);
        #pragma unroll
        for (int c = 0; c < ENCIN_; ++c)
            acc += x_enc[(b * S_ + j) * ENCIN_ + c] * tk[(w * ENCIN_ + c) * D_ + dm];
    }
    int   i2    = dm & ~1;
    float freq  = __expf(-(float)i2 * (logf(10000.f) / (float)D_));
    float angle = (float)s * freq;
    float pe    = (dm & 1) ? cosf(angle) : sinf(angle);
    X[idx] = acc + pe;
}

// ---------------------------------------------------------------------------
// WMMA GEMM:  C[M,N] = A[M,K] @ W[N,K]^T + bias   (f16 in, fp32 acc)
// block tile 128x64, 8 waves (4 M-strips x 2 N-strips), wave tile 32x32
// TDM double-buffered LDS staging when available.
// act: 0 = none, 1 = exact GELU.  Ch != null -> f16 output, else fp32 to C.
// ---------------------------------------------------------------------------
__global__ __launch_bounds__(256)
void gemm_wmma_kernel(const _Float16* __restrict__ A, const _Float16* __restrict__ W,
                      const float* __restrict__ bias, float* __restrict__ C,
                      _Float16* __restrict__ Ch,
                      int M, int N, int K, int act)
{
    __shared__ _Float16 As[2][128 * 32];   // [buf][m][k]
    __shared__ _Float16 Bs[2][64 * 32];    // [buf][n][k]

    const int tid  = threadIdx.x;
    const int lane = tid & 31;
    const int wave = tid >> 5;
    const int wm   = wave & 3;
    const int wn   = wave >> 2;
    const int blockM = blockIdx.y * 128;
    const int blockN = blockIdx.x * 64;
    const int nsteps = K >> 5;

    v8f acc[2][2] = {};

    const int mrow    = lane & 15;
    const int kShiftA = (lane >> 4) * 8;
    const int kBaseB  = (lane >> 4) * 16;

#if USE_TDM
    if (wave == 0) {
        tdm_load_tile(A + (size_t)blockM * K, (u32)(uintptr_t)&As[0][0], 32, 128, (u32)K);
        tdm_load_tile(W + (size_t)blockN * K, (u32)(uintptr_t)&Bs[0][0], 32, 64,  (u32)K);
    }
#endif

    for (int step = 0; step < nsteps; ++step) {
        const int buf = step & 1;
#if USE_TDM
        if (wave == 0) {
            if (step + 1 < nsteps) {
                const size_t k0n = (size_t)(step + 1) * 32;
                tdm_load_tile(A + (size_t)blockM * K + k0n,
                              (u32)(uintptr_t)&As[buf ^ 1][0], 32, 128, (u32)K);
                tdm_load_tile(W + (size_t)blockN * K + k0n,
                              (u32)(uintptr_t)&Bs[buf ^ 1][0], 32, 64, (u32)K);
                __builtin_amdgcn_s_wait_tensorcnt(2);   // current buf complete
            } else {
                __builtin_amdgcn_s_wait_tensorcnt(0);
            }
        }
        __syncthreads();
#else
        const int k0 = step * 32;
        #pragma unroll
        for (int j = 0; j < 16; ++j) {
            int li = tid * 16 + j;
            int r = li >> 5, c = li & 31;
            As[buf][li] = A[(size_t)(blockM + r) * K + (k0 + c)];
        }
        #pragma unroll
        for (int j = 0; j < 8; ++j) {
            int li = tid * 8 + j;
            int n = li >> 5, c = li & 31;
            Bs[buf][li] = W[(size_t)(blockN + n) * K + (k0 + c)];
        }
        __syncthreads();
#endif

        #pragma unroll
        for (int mi = 0; mi < 2; ++mi) {
            v16h af;   // A fragment 16x32 (ISA 7.12.2 16-bit A layout)
            const int mb = wm * 32 + mi * 16 + mrow;
            #pragma unroll
            for (int e = 0; e < 16; ++e) {
                int kk = e + ((e >= 8) ? 8 : 0) + kShiftA;
                af[e] = As[buf][mb * 32 + kk];
            }
            #pragma unroll
            for (int ni = 0; ni < 2; ++ni) {
                v16h bf;   // B fragment 32x16
                const int nb = wn * 32 + ni * 16 + (lane & 15);
                #pragma unroll
                for (int e = 0; e < 16; ++e)
                    bf[e] = Bs[buf][nb * 32 + e + ((e >= 8) ? 8 : 0) + kShiftA];
                acc[mi][ni] = __builtin_amdgcn_wmma_f32_16x16x32_f16(
                    false, af, false, bf, (short)0, acc[mi][ni], false, false);
            }
        }
        __syncthreads();   // all waves done with buf before it is refilled
    }

    const int rowOff = (lane >> 4) * 8;
    const int col    = lane & 15;
    #pragma unroll
    for (int mi = 0; mi < 2; ++mi) {
        #pragma unroll
        for (int ni = 0; ni < 2; ++ni) {
            int n  = blockN + wn * 32 + ni * 16 + col;
            float bv = bias ? bias[n] : 0.f;
            #pragma unroll
            for (int r = 0; r < 8; ++r) {
                int m = blockM + wm * 32 + mi * 16 + rowOff + r;
                float v = acc[mi][ni][r] + bv;
                if (act == 1) v = 0.5f * v * (1.f + erff(v * 0.70710678118f));
                if (Ch) Ch[(size_t)m * N + n] = (_Float16)v;
                else    C [(size_t)m * N + n] = v;
            }
        }
    }
}

// ---------------------------------------------------------------------------
// ProbSparse attention pieces (fp32 VALU; tiny share of total FLOPs)
// ---------------------------------------------------------------------------
__global__ void gen_idx_kernel(int* __restrict__ idx, unsigned seed)
{
    int t = blockIdx.x * blockDim.x + threadIdx.x;
    if (t >= S_ * NTOP_) return;
    unsigned h = (unsigned)t * 2654435761u ^ seed;
    h ^= h >> 16; h *= 0x7feb352du;
    h ^= h >> 15; h *= 0x846ca68bu;
    h ^= h >> 16;
    idx[t] = (int)(h & (S_ - 1));
}

__global__ void sample_m_kernel(const float* __restrict__ Q,
                                const float* __restrict__ Kb,
                                const int* __restrict__ idx,
                                float* __restrict__ Mbuf)
{
    int t = blockIdx.x * blockDim.x + threadIdx.x;   // over B*H*S
    if (t >= B_ * H_ * S_) return;
    int l = t % S_;
    int h = (t / S_) % H_;
    int b = t / (S_ * H_);
    const float* qrow = Q + (size_t)(b * S_ + l) * D_ + h * DH_;
    float mx = -INFINITY, sm = 0.f;
    for (int s = 0; s < NTOP_; ++s) {
        int kr = idx[l * NTOP_ + s];
        const float* krow = Kb + (size_t)(b * S_ + kr) * D_ + h * DH_;
        float d = 0.f;
        #pragma unroll 8
        for (int e = 0; e < DH_; ++e) d += qrow[e] * krow[e];
        mx = fmaxf(mx, d);
        sm += d;
    }
    Mbuf[t] = mx - sm / (float)S_;
}

__global__ void topk_kernel(const float* __restrict__ Mbuf, int* __restrict__ topk)
{
    __shared__ float Ms[S_];
    int bh   = blockIdx.x;
    int lane = threadIdx.x;
    for (int i = lane; i < S_; i += 32) Ms[i] = Mbuf[(size_t)bh * S_ + i];
    __syncthreads();
    for (int u = 0; u < NTOP_; ++u) {
        float bv = -INFINITY; int bi = 0;
        for (int i = lane; i < S_; i += 32) {
            float v = Ms[i];
            if (v > bv) { bv = v; bi = i; }
        }
        #pragma unroll
        for (int off = 16; off > 0; off >>= 1) {
            float ov = __shfl_xor(bv, off, 32);
            int   oi = __shfl_xor(bi, off, 32);
            if (ov > bv || (ov == bv && oi < bi)) { bv = ov; bi = oi; }
        }
        if (lane == 0) { topk[bh * NTOP_ + u] = bi; Ms[bi] = -INFINITY; }
        __syncthreads();
    }
}

__global__ void ctx_mean_kernel(const float* __restrict__ V, float* __restrict__ T)
{
    __shared__ float red[4][DH_];
    int bh = blockIdx.x;
    int b = bh / H_, h = bh % H_;
    int t = threadIdx.x;
    int d = t & 63, chunk = t >> 6;
    float acc = 0.f;
    for (int l = chunk; l < S_; l += 4)
        acc += V[(size_t)(b * S_ + l) * D_ + h * DH_ + d];
    red[chunk][d] = acc;
    __syncthreads();
    if (chunk == 0)
        red[0][d] = (red[0][d] + red[1][d] + red[2][d] + red[3][d]) / (float)S_;
    __syncthreads();
    float m = red[0][d];
    for (int l = chunk; l < S_; l += 4)
        T[(size_t)(b * S_ + l) * D_ + h * DH_ + d] = m;
}

__global__ __launch_bounds__(256)
void topq_kernel(const float* __restrict__ Q, const float* __restrict__ Kb,
                 const float* __restrict__ V, const int* __restrict__ topk,
                 float* __restrict__ T)
{
    __shared__ float sc[S_];
    __shared__ float red[256];
    __shared__ float qs[DH_];
    int bid = blockIdx.x;
    int u = bid % NTOP_;
    int h = (bid / NTOP_) % H_;
    int b = bid / (NTOP_ * H_);
    int t = threadIdx.x;
    int qi = topk[(b * H_ + h) * NTOP_ + u];

    if (t < DH_) qs[t] = Q[(size_t)(b * S_ + qi) * D_ + h * DH_ + t];
    __syncthreads();

    for (int k = t; k < S_; k += 256) {
        const float* krow = Kb + (size_t)(b * S_ + k) * D_ + h * DH_;
        float d = 0.f;
        #pragma unroll 8
        for (int e = 0; e < DH_; ++e) d += qs[e] * krow[e];
        sc[k] = d * 0.125f;
    }
    __syncthreads();

    float lm = -INFINITY;
    for (int k = t; k < S_; k += 256) lm = fmaxf(lm, sc[k]);
    red[t] = lm; __syncthreads();
    for (int off = 128; off > 0; off >>= 1) {
        if (t < off) red[t] = fmaxf(red[t], red[t + off]);
        __syncthreads();
    }
    float mx = red[0];
    __syncthreads();

    float ls = 0.f;
    for (int k = t; k < S_; k += 256) {
        float e = __expf(sc[k] - mx);
        sc[k] = e; ls += e;
    }
    red[t] = ls; __syncthreads();
    for (int off = 128; off > 0; off >>= 1) {
        if (t < off) red[t] += red[t + off];
        __syncthreads();
    }
    float inv = 1.f / red[0];

    if (t < DH_) {
        float acc = 0.f;
        for (int k = 0; k < S_; ++k)
            acc += sc[k] * V[(size_t)(b * S_ + k) * D_ + h * DH_ + t];
        T[(size_t)(b * S_ + qi) * D_ + h * DH_ + t] = acc * inv;
    }
}

// ---------------------------------------------------------------------------
// LayerNorm with optional fused residual
// ---------------------------------------------------------------------------
__global__ __launch_bounds__(256)
void layernorm_kernel(const float* __restrict__ x, const float* __restrict__ res,
                      const float* __restrict__ g, const float* __restrict__ bta,
                      float* __restrict__ out)
{
    __shared__ float red[256];
    __shared__ float row[D_];
    size_t r = blockIdx.x;
    int t = threadIdx.x;

    float s = 0.f;
    for (int d = t; d < D_; d += 256) {
        float v = x[r * D_ + d] + (res ? res[r * D_ + d] : 0.f);
        row[d] = v; s += v;
    }
    red[t] = s; __syncthreads();
    for (int off = 128; off > 0; off >>= 1) {
        if (t < off) red[t] += red[t + off];
        __syncthreads();
    }
    float mu = red[0] / (float)D_;
    __syncthreads();

    float v2 = 0.f;
    for (int d = t; d < D_; d += 256) { float dv = row[d] - mu; v2 += dv * dv; }
    red[t] = v2; __syncthreads();
    for (int off = 128; off > 0; off >>= 1) {
        if (t < off) red[t] += red[t + off];
        __syncthreads();
    }
    float inv = rsqrtf(red[0] / (float)D_ + EPS_);

    for (int d = t; d < D_; d += 256)
        out[r * D_ + d] = (row[d] - mu) * inv * g[d] + bta[d];
}

__global__ void gelu_mark_kernel(const float* __restrict__ xin,
                                 const float* __restrict__ mark,
                                 float* __restrict__ out)
{
    int idx = blockIdx.x * blockDim.x + threadIdx.x;
    if (idx >= B_ * S_ * D_) return;
    int s = (idx / D_) % S_;
    int b = idx / (D_ * S_);
    float v = xin[idx];
    v = 0.5f * v * (1.f + erff(v * 0.70710678118f));
    out[idx] = v * mark[b * S_ + s];
}

__global__ __launch_bounds__(256)
void proj_kernel(const float* __restrict__ act, const float* __restrict__ pw,
                 const float* __restrict__ pb, float* __restrict__ out)
{
    __shared__ float red[256];
    int bc = blockIdx.x;
    int c = bc % NCLS_, b = bc / NCLS_;
    int t = threadIdx.x;
    const size_t N = (size_t)S_ * D_;
    const float* a = act + (size_t)b * N;
    const float* w = pw + (size_t)c * N;
    float s = 0.f;
    for (size_t i = t; i < N; i += 256) s += a[i] * w[i];
    red[t] = s; __syncthreads();
    for (int off = 128; off > 0; off >>= 1) {
        if (t < off) red[t] += red[t + off];
        __syncthreads();
    }
    if (t == 0) out[b * NCLS_ + c] = red[0] + pb[c];
}

// ---------------------------------------------------------------------------
extern "C" void kernel_launch(void* const* d_in, const int* in_sizes, int n_in,
                              void* d_out, int out_size, void* d_ws, size_t ws_size,
                              hipStream_t stream)
{
    (void)in_sizes; (void)n_in; (void)out_size; (void)ws_size;
    const float* x_enc  = (const float*)d_in[0];
    const float* x_mark = (const float*)d_in[1];
    const float* tk     = (const float*)d_in[2];
    const float* Wq = (const float*)d_in[3];  const float* bq = (const float*)d_in[4];
    const float* Wk = (const float*)d_in[5];  const float* bk = (const float*)d_in[6];
    const float* Wv = (const float*)d_in[7];  const float* bv = (const float*)d_in[8];
    const float* Wo = (const float*)d_in[9];  const float* bo = (const float*)d_in[10];
    const float* c1w = (const float*)d_in[11]; const float* c1b = (const float*)d_in[12];
    const float* c2w = (const float*)d_in[13]; const float* c2b = (const float*)d_in[14];
    const float* g1 = (const float*)d_in[15]; const float* b1 = (const float*)d_in[16];
    const float* g2 = (const float*)d_in[17]; const float* b2 = (const float*)d_in[18];
    const float* gf = (const float*)d_in[19]; const float* bf = (const float*)d_in[20];
    const float* proj_w = (const float*)d_in[21];
    const float* proj_b = (const float*)d_in[22];

    const size_t SD = (size_t)B_ * S_ * D_;          // 4,194,304
    float* X  = (float*)d_ws;                        // activations
    float* T  = X + SD;                              // attn ctx / ffn out
    float* Qb = X + 2 * SD;
    float* Kb = X + 3 * SD;
    float* Vb = X + 4 * SD;
    float* Mb = X + 5 * SD;                          // B*H*S floats
    int*  idxb  = (int*)(Mb + (size_t)B_ * H_ * S_);
    int*  topkb = idxb + S_ * NTOP_;
    _Float16* AH = (_Float16*)(topkb + B_ * H_ * NTOP_); // f16 activations, SD halves
    _Float16* BH = AH + SD;                              // f16 ffn hidden, B*S*DFF halves
    _Float16* WH = BH + (size_t)B_ * S_ * DFF_;          // f16 weights, up to DFF*D halves

    const int rows = B_ * S_;                        // 8192
    const int cvtB = 256;
    #define CVT(src, dst, n) \
        f32_to_f16_kernel<<<((n) + cvtB - 1) / cvtB, cvtB, 0, stream>>>(src, dst, n)

    token_embed_kernel<<<(B_ * S_ * D_ + 255) / 256, 256, 0, stream>>>(x_enc, tk, X);

    for (int L = 0; L < NL_; ++L) {
        const float* wqL = Wq + (size_t)L * D_ * D_;   const float* bqL = bq + L * D_;
        const float* wkL = Wk + (size_t)L * D_ * D_;   const float* bkL = bk + L * D_;
        const float* wvL = Wv + (size_t)L * D_ * D_;   const float* bvL = bv + L * D_;
        const float* woL = Wo + (size_t)L * D_ * D_;   const float* boL = bo + L * D_;
        const float* c1wL = c1w + (size_t)L * DFF_ * D_; const float* c1bL = c1b + L * DFF_;
        const float* c2wL = c2w + (size_t)L * D_ * DFF_; const float* c2bL = c2b + L * D_;
        const float* g1L = g1 + L * D_; const float* b1L = b1 + L * D_;
        const float* g2L = g2 + L * D_; const float* b2L = b2 + L * D_;

        dim3 gD(D_ / 64, rows / 128);                  // (8, 64)
        CVT(X, AH, (int)SD);
        CVT(wqL, WH, D_ * D_);
        gemm_wmma_kernel<<<gD, 256, 0, stream>>>(AH, WH, bqL, Qb, nullptr, rows, D_, D_, 0);
        CVT(wkL, WH, D_ * D_);
        gemm_wmma_kernel<<<gD, 256, 0, stream>>>(AH, WH, bkL, Kb, nullptr, rows, D_, D_, 0);
        CVT(wvL, WH, D_ * D_);
        gemm_wmma_kernel<<<gD, 256, 0, stream>>>(AH, WH, bvL, Vb, nullptr, rows, D_, D_, 0);

        gen_idx_kernel<<<(S_ * NTOP_ + 255) / 256, 256, 0, stream>>>(
            idxb, 0x9E3779B9u * (unsigned)(L + 1));
        sample_m_kernel<<<(B_ * H_ * S_) / 256, 256, 0, stream>>>(Qb, Kb, idxb, Mb);
        topk_kernel<<<B_ * H_, 32, 0, stream>>>(Mb, topkb);
        ctx_mean_kernel<<<B_ * H_, 256, 0, stream>>>(Vb, T);
        topq_kernel<<<B_ * H_ * NTOP_, 256, 0, stream>>>(Qb, Kb, Vb, topkb, T);

        // merged-head output projection -> Qb (QKV dead now)
        CVT(T, AH, (int)SD);
        CVT(woL, WH, D_ * D_);
        gemm_wmma_kernel<<<gD, 256, 0, stream>>>(AH, WH, boL, Qb, nullptr, rows, D_, D_, 0);
        layernorm_kernel<<<rows, 256, 0, stream>>>(X, Qb, g1L, b1L, X);

        // FFN: hidden kept in f16 (it is a WMMA A-operand for FFN2)
        CVT(X, AH, (int)SD);
        CVT(c1wL, WH, DFF_ * D_);
        dim3 gF1(DFF_ / 64, rows / 128);
        gemm_wmma_kernel<<<gF1, 256, 0, stream>>>(AH, WH, c1bL, nullptr, BH, rows, DFF_, D_, 1);
        CVT(c2wL, WH, D_ * DFF_);
        dim3 gF2(D_ / 64, rows / 128);
        gemm_wmma_kernel<<<gF2, 256, 0, stream>>>(BH, WH, c2bL, T, nullptr, rows, D_, DFF_, 0);
        layernorm_kernel<<<rows, 256, 0, stream>>>(X, T, g2L, b2L, X);
    }

    layernorm_kernel<<<rows, 256, 0, stream>>>(X, nullptr, gf, bf, T);
    gelu_mark_kernel<<<(B_ * S_ * D_ + 255) / 256, 256, 0, stream>>>(T, x_mark, T);
    proj_kernel<<<B_ * NCLS_, 256, 0, stream>>>(T, proj_w, proj_b, (float*)d_out);
    #undef CVT
}
